// GlobalAttention_72627896975609
// MI455X (gfx1250) — compile-verified, matrix operand order checked by reading, not run on hardware
//
#include <hip/hip_runtime.h>
#include <math.h>

// ---- problem constants (from reference) ----
#define Bdim 1
#define Rdim 512
#define Sdim 2048
#define Cdim 64
#define Hdim 8
#define HDdim 8
#define INFV 1e9f
#define EPSV 1e-10f

typedef __attribute__((ext_vector_type(16))) __bf16 v16bf;
typedef __attribute__((ext_vector_type(8)))  float  v8f;
typedef __attribute__((ext_vector_type(4)))  int    v4i;

// ---- dynamic LDS layout (bytes) ----
static constexpr unsigned OFF_LOGIT = 0;                          // [8][2048] f32 = 65536
static constexpr unsigned OFF_POOL  = OFF_LOGIT + 8 * Sdim * 4;   // [4][64] f32  = 1024
static constexpr unsigned OFF_QP    = OFF_POOL  + 4 * 64 * 4;     // [64] f32     = 256
static constexpr unsigned OFF_WQH   = OFF_QP    + 64 * 4;         // [8][64] f32  = 2048
static constexpr unsigned OFF_PM    = OFF_WQH   + 8 * 64 * 4;     // [8][64] f32  = 2048
static constexpr unsigned OFF_O     = OFF_PM    + 8 * 64 * 4;     // [64] f32     = 256
static constexpr unsigned OFF_MS    = OFF_O     + 64 * 4;         // [4] f32 +pad = 64
static constexpr unsigned OFF_TBUF  = OFF_MS    + 64;             // 8*16*64 bf16 = 16384
static constexpr unsigned OFF_STAGE = OFF_TBUF  + 8 * 16 * 64 * 2;// 8 waves * 2 * 4KB f32 tiles
static constexpr unsigned SMEM_BYTES = OFF_STAGE + 8 * 2 * 16 * 64 * 4;  // 153,152 B < 320 KB/WGP

// ---- async global->LDS copy (ASYNCcnt path), with safe fallback ----
#if defined(__HIP_DEVICE_COMPILE__) && __has_builtin(__builtin_amdgcn_global_load_async_to_lds_b128)
#define HAS_ASYNC_LDS 1
#else
#define HAS_ASYNC_LDS 0
#endif

#if HAS_ASYNC_LDS
typedef v4i __attribute__((address_space(1))) g_v4i;   // global (AS1) 4xi32
typedef v4i __attribute__((address_space(3))) l_v4i;   // LDS    (AS3) 4xi32
#endif

template <int N>
static __device__ inline void wait_async() {
#if defined(__HIP_DEVICE_COMPILE__) && __has_builtin(__builtin_amdgcn_s_wait_asynccnt)
    __builtin_amdgcn_s_wait_asynccnt((unsigned short)N);
#elif HAS_ASYNC_LDS
    asm volatile("s_wait_asynccnt %0" :: "n"(N) : "memory");
#endif
    asm volatile("" ::: "memory");
}

// Copy one 16x64 f32 tile (4 KB) global -> LDS. 32 lanes x 8 x b128.
// ldsOff: byte offset of destination within LDS; gdst: generic pointer to same (fallback path).
static __device__ inline void tile_copy_async(unsigned ldsOff, float* gdst,
                                              const float* src, int lane) {
#if HAS_ASYNC_LDS
#pragma unroll
    for (int i = 0; i < 8; ++i) {
        g_v4i* gs = (g_v4i*)(size_t)(src + lane * 32 + i * 4);
        l_v4i* ls = (l_v4i*)(size_t)(ldsOff + lane * 128 + i * 16);
        __builtin_amdgcn_global_load_async_to_lds_b128(gs, ls, 0, 0);
    }
#else
    const float4* s = (const float4*)src + lane * 8;
    float4*       d = (float4*)gdst + lane * 8;
#pragma unroll
    for (int i = 0; i < 8; ++i) d[i] = s[i];
#endif
}

// D = A(16x32 bf16) * B(32x16 bf16) + C(16x16 f32)
static __device__ inline v8f wmma_bf16(v16bf a, v16bf b, v8f c) {
    return __builtin_amdgcn_wmma_f32_16x16x32_bf16(
        false, a, false, b, (short)0, c, false, false);
}

// A fragment from a staged f32 LDS tile (16 x 64, row-major).
static __device__ inline v16bf frag_a_stage(const float* tile, int k0, int lane) {
    const float* rp = tile + (lane & 15) * 64 + k0 + ((lane >> 4) << 3);
    v16bf f;
#pragma unroll
    for (int e = 0; e < 8; ++e) f[e] = (__bf16)rp[e];
#pragma unroll
    for (int e = 0; e < 8; ++e) f[e + 8] = (__bf16)rp[16 + e];
    return f;
}

// A fragment from a bf16 row-major LDS tile (16 x 64).
static __device__ inline v16bf frag_a_lds(const __bf16* tb, int k0, int lane) {
    const __bf16* rp = tb + (lane & 15) * 64 + k0 + ((lane >> 4) << 3);
    v16bf f;
#pragma unroll
    for (int e = 0; e < 8; ++e) f[e] = rp[e];
#pragma unroll
    for (int e = 0; e < 8; ++e) f[e + 8] = rp[16 + e];
    return f;
}

// B fragment: 32 K x 16 cols from row-major f32 matrix (rows = K, ld = row stride).
static __device__ inline v16bf frag_b_f32(const float* mat, int ld, int k0, int n0, int lane) {
    const float* cp = mat + (size_t)(k0 + ((lane >> 4) << 3)) * ld + n0 + (lane & 15);
    v16bf f;
#pragma unroll
    for (int e = 0; e < 8; ++e) f[e] = (__bf16)cp[e * ld];
#pragma unroll
    for (int e = 0; e < 8; ++e) f[e + 8] = (__bf16)cp[(16 + e) * ld];
    return f;
}

// A fragment for probabilities p[8][2048] in LDS f32; rows 8..15 are zero.
static __device__ inline v16bf frag_a_probs(const float* p, int k0, int lane) {
    v16bf f;
    int row = lane & 15;
    if (row < 8) {
        const float* rp = p + (size_t)row * Sdim + k0 + ((lane >> 4) << 3);
#pragma unroll
        for (int e = 0; e < 8; ++e) f[e] = (__bf16)rp[e];
#pragma unroll
        for (int e = 0; e < 8; ++e) f[e + 8] = (__bf16)rp[16 + e];
    } else {
#pragma unroll
        for (int e = 0; e < 16; ++e) f[e] = (__bf16)0.0f;
    }
    return f;
}

__global__ void __launch_bounds__(256)
ga_kernel(const float* __restrict__ m, const float* __restrict__ mask,
          const float* __restrict__ Wq, const float* __restrict__ Wk,
          const float* __restrict__ Wv, const float* __restrict__ Wg,
          const float* __restrict__ bg, const float* __restrict__ Wo,
          const float* __restrict__ bo, float* __restrict__ out)
{
    const int r    = blockIdx.x;          // one workgroup per (b, r)
    const int tid  = threadIdx.x;         // 256 threads = 8 wave32
    const int lane = tid & 31;
    const int wave = tid >> 5;

    extern __shared__ char smem[];
    float*  s_logit = (float*)(smem + OFF_LOGIT);
    float*  s_pool  = (float*)(smem + OFF_POOL);
    float*  s_qp    = (float*)(smem + OFF_QP);
    float*  s_wqh   = (float*)(smem + OFF_WQH);
    float*  s_pm    = (float*)(smem + OFF_PM);
    float*  s_o     = (float*)(smem + OFF_O);
    float*  s_ms    = (float*)(smem + OFF_MS);
    __bf16* s_tbuf  = (__bf16*)(smem + OFF_TBUF);

    const float* mr    = m    + (size_t)r * Sdim * Cdim;
    const float* maskr = mask + (size_t)r * Sdim;
    float*       outr  = out  + (size_t)r * Sdim * Cdim;

    // Per-wave double-buffered staging tiles (16x64 f32 each).
    float*   stage_g   = (float*)(smem + OFF_STAGE) + wave * 2048;
    unsigned stage_off = OFF_STAGE + wave * 8192;

    // Kick off the async DMA of this wave's FIRST phase-3 tile right now:
    // it overlaps the entire pool/logits/softmax prologue.
    tile_copy_async(stage_off, stage_g, mr + (size_t)(wave * 16) * Cdim, lane);

    // ---------- Phase 1: masked mean pool over S ----------
    {
        const int c = tid & 63, grp = tid >> 6;
        float acc = 0.0f, msum = 0.0f;
        for (int s = grp; s < Sdim; s += 4) {
            float mk = maskr[s];
            acc += mr[(size_t)s * Cdim + c] * mk;
            if (c == 0) msum += mk;
        }
        s_pool[grp * 64 + c] = acc;
        if (c == 0) s_ms[grp] = msum;
    }
    __syncthreads();
    if (tid < 64) {
        float p  = s_pool[tid] + s_pool[64 + tid] + s_pool[128 + tid] + s_pool[192 + tid];
        float ms = s_ms[0] + s_ms[1] + s_ms[2] + s_ms[3];
        s_qp[tid] = p / (ms + EPSV);
    }
    __syncthreads();
    if (tid < 64) {   // q = (qpool @ Wq) * HD^-0.5
        float qj = 0.0f;
        for (int c = 0; c < Cdim; ++c) qj += s_qp[c] * Wq[c * 64 + tid];
        s_pool[tid] = qj * 0.35355339059327373f;
    }
    __syncthreads();
    for (int i = tid; i < Hdim * Cdim; i += 256) {   // wqh[h][c] = sum_d Wk[c][d]*q[h][d]
        int h = i >> 6, c = i & 63;
        float v = 0.0f;
#pragma unroll
        for (int d = 0; d < HDdim; ++d) v += Wk[c * HDdim + d] * s_pool[h * HDdim + d];
        s_wqh[h * 64 + c] = v;
    }
    __syncthreads();

    // ---------- Phase 2: logits = m @ wqh^T + mask bias ----------
    for (int s = tid; s < Sdim; s += 256) {
        const float* mrow = mr + (size_t)s * Cdim;
        float bias = INFV * (maskr[s] - 1.0f);
        float l[Hdim];
#pragma unroll
        for (int h = 0; h < Hdim; ++h) l[h] = bias;
        for (int c = 0; c < Cdim; ++c) {
            float mv = mrow[c];
#pragma unroll
            for (int h = 0; h < Hdim; ++h) l[h] += mv * s_wqh[h * 64 + c];
        }
#pragma unroll
        for (int h = 0; h < Hdim; ++h) s_logit[h * Sdim + s] = l[h];
    }
    __syncthreads();

    // ---------- Phase 2b: softmax over S, wave w = head w ----------
    {
        const int h = wave;
        float mx = -3.4e38f;
        for (int s = lane; s < Sdim; s += 32) mx = fmaxf(mx, s_logit[h * Sdim + s]);
#pragma unroll
        for (int off = 16; off > 0; off >>= 1) mx = fmaxf(mx, __shfl_xor(mx, off, 32));
        float sum = 0.0f;
        for (int s = lane; s < Sdim; s += 32) {
            float e = __expf(s_logit[h * Sdim + s] - mx);
            s_logit[h * Sdim + s] = e;
            sum += e;
        }
#pragma unroll
        for (int off = 16; off > 0; off >>= 1) sum += __shfl_xor(sum, off, 32);
        float inv = 1.0f / sum;
        for (int s = lane; s < Sdim; s += 32) s_logit[h * Sdim + s] *= inv;
    }
    __syncthreads();

    // ---------- Phase 2c: pm = p @ m (8x64, K=2048) via WMMA, waves 0..3 ----------
    if (wave < 4) {
        const int nt = wave;
        v8f acc = {};
        for (int kc = 0; kc < Sdim / 32; ++kc) {
            v16bf ap = frag_a_probs(s_logit, kc * 32, lane);
            v16bf bm = frag_b_f32(mr, Cdim, kc * 32, nt * 16, lane);
            acc = wmma_bf16(ap, bm, acc);
        }
        if (lane < 16) {
            int c = nt * 16 + lane;
#pragma unroll
            for (int i = 0; i < 8; ++i) s_pm[i * 64 + c] = acc[i];
        }
    }
    __syncthreads();
    if (tid < 64) {   // o[h*8+d] = pm[h] @ Wv
        int h = tid >> 3, d = tid & 7;
        float v = 0.0f;
        for (int c = 0; c < Cdim; ++c) v += s_pm[h * 64 + c] * Wv[c * HDdim + d];
        s_o[tid] = v;
    }
    __syncthreads();

    // ---------- Phase 3: out = (sigmoid(m@Wg + bg) * o) @ Wo + bo ----------
    v16bf BWg[2][4], BWo[2][4];
#pragma unroll
    for (int kf = 0; kf < 2; ++kf)
#pragma unroll
        for (int nt = 0; nt < 4; ++nt) {
            BWg[kf][nt] = frag_b_f32(Wg, 64, kf * 32, nt * 16, lane);
            BWo[kf][nt] = frag_b_f32(Wo, 64, kf * 32, nt * 16, lane);
        }
    float ov[4], bgv[4], bov[4];
#pragma unroll
    for (int nt = 0; nt < 4; ++nt) {
        int j   = nt * 16 + (lane & 15);
        ov[nt]  = s_o[j];
        bgv[nt] = bg[j];
        bov[nt] = bo[j];
    }
    __bf16* tb = s_tbuf + wave * 16 * 64;
    const int n = lane & 15, half = lane >> 4;

    int buf = 0;
    for (int st = wave; st < Sdim / 16; st += 8) {
        const int s0 = st << 4;

        // Software pipeline: issue DMA for tile st+8, then drain tile st.
        if (st + 8 < Sdim / 16) {
            tile_copy_async(stage_off + (buf ^ 1) * 4096, stage_g + (buf ^ 1) * 1024,
                            mr + (size_t)(s0 + 128) * Cdim, lane);
            wait_async<8>();     // current tile's 8 b128 copies retired
        } else {
            wait_async<0>();
        }
        const float* cur = stage_g + buf * 1024;

        v16bf A0 = frag_a_stage(cur, 0, lane);
        v16bf A1 = frag_a_stage(cur, 32, lane);

        // GEMM 1: G = m @ Wg, sigmoid-gate with o, stage bf16 A-tile in LDS
#pragma unroll
        for (int nt = 0; nt < 4; ++nt) {
            v8f acc = {};
            acc = wmma_bf16(A0, BWg[0][nt], acc);
            acc = wmma_bf16(A1, BWg[1][nt], acc);
            float oj = ov[nt], bgj = bgv[nt];
#pragma unroll
            for (int i = 0; i < 8; ++i) {
                float x = acc[i] + bgj;
                float t = oj / (1.0f + __expf(-x));       // sigmoid(x) * o_j
                tb[(i + 8 * half) * 64 + nt * 16 + n] = (__bf16)t;
            }
        }

        // GEMM 2: out = t @ Wo + bo   (non-temporal stores: out is write-once,
        // keep L2 for the re-read m slabs)
        v16bf T0 = frag_a_lds(tb, 0, lane);
        v16bf T1 = frag_a_lds(tb, 32, lane);
#pragma unroll
        for (int nt = 0; nt < 4; ++nt) {
            v8f acc = {};
            acc = wmma_bf16(T0, BWo[0][nt], acc);
            acc = wmma_bf16(T1, BWo[1][nt], acc);
            float boj = bov[nt];
#pragma unroll
            for (int i = 0; i < 8; ++i) {
                int srow = s0 + i + 8 * half;
                __builtin_nontemporal_store(acc[i] + boj,
                                            &outr[(size_t)srow * Cdim + nt * 16 + n]);
            }
        }
        buf ^= 1;
    }
}

extern "C" void kernel_launch(void* const* d_in, const int* in_sizes, int n_in,
                              void* d_out, int out_size, void* d_ws, size_t ws_size,
                              hipStream_t stream) {
    const float* m    = (const float*)d_in[0];
    const float* mask = (const float*)d_in[1];
    const float* Wq   = (const float*)d_in[2];
    const float* Wk   = (const float*)d_in[3];
    const float* Wv   = (const float*)d_in[4];
    const float* Wg   = (const float*)d_in[5];
    const float* bg   = (const float*)d_in[6];
    const float* Wo   = (const float*)d_in[7];
    const float* bo   = (const float*)d_in[8];
    float* out = (float*)d_out;

    (void)hipFuncSetAttribute((const void*)ga_kernel,
                              hipFuncAttributeMaxDynamicSharedMemorySize,
                              (int)SMEM_BYTES);

    ga_kernel<<<Bdim * Rdim, 256, SMEM_BYTES, stream>>>(
        m, mask, Wq, Wk, Wv, Wg, bg, Wo, bo, out);
    (void)in_sizes; (void)n_in; (void)out_size; (void)d_ws; (void)ws_size;
}
